// ConceptLayer_22514218566337
// MI455X (gfx1250) — compile-verified, matrix operand order checked by reading, not run on hardware
//
#include <hip/hip_runtime.h>

typedef __attribute__((ext_vector_type(16))) __bf16          v16bf;
typedef __attribute__((ext_vector_type(16))) unsigned short  v16us;
typedef __attribute__((ext_vector_type(8)))  float           v8f;

#define S_LEN 512
#define E_DIM 256
#define T_TOK 1024      /* B*S */
#define K_TOT 65536     /* E*E */

__device__ __forceinline__ unsigned short f2bfu(float f) {
  unsigned u = __builtin_bit_cast(unsigned, f);
  return (unsigned short)((u + 0x7FFFu + ((u >> 16) & 1u)) >> 16);   // RNE
}
__device__ __forceinline__ float bfu2f(unsigned short h) {
  return __builtin_bit_cast(float, ((unsigned)h) << 16);
}
// packed bf16 multiply (VOP3P V_PK_MUL_BF16): two bf16 products per op
__device__ __forceinline__ unsigned pk_mul_bf16(unsigned a, unsigned b) {
  unsigned d;
  asm("v_pk_mul_bf16 %0, %1, %2" : "=v"(d) : "v"(a), "v"(b));
  return d;
}

// ---------------------------------------------------------------------------
// Kernel 1: s[b,i,e] = sum_{j<i} x[b,j,e]/(i-j)^2 ; also bf16 copies of x, s.
// ---------------------------------------------------------------------------
__global__ void prefix_kernel(const float* __restrict__ x,
                              unsigned short* __restrict__ x_bf,
                              unsigned short* __restrict__ s_bf) {
  int idx = blockIdx.x * 256 + threadIdx.x;   // (b*S + i)*E + e
  int e = idx & (E_DIM - 1);
  int t = idx >> 8;
  int i = t & (S_LEN - 1);
  int b = t >> 9;
  const float* xb = x + (size_t)b * S_LEN * E_DIM + e;
  float acc = 0.f;
  for (int d = 1; d <= i; ++d) {
    float fd = (float)d;
    acc += xb[(i - d) * E_DIM] * __builtin_amdgcn_rcpf(fd * fd);
  }
  x_bf[idx] = f2bfu(x[idx]);
  s_bf[idx] = f2bfu(acc);
}

// ---------------------------------------------------------------------------
// Kernel 2: concept_map fp32 -> bf16 (rounded), 8 elements / thread.
// ---------------------------------------------------------------------------
__global__ void cvt_cm_kernel(const float* __restrict__ cm,
                              unsigned short* __restrict__ cm_bf) {
  size_t i = ((size_t)blockIdx.x * 256 + threadIdx.x) * 8;
  float4 a = *reinterpret_cast<const float4*>(cm + i);
  float4 b = *reinterpret_cast<const float4*>(cm + i + 4);
  uint4 o;
  o.x = (unsigned)f2bfu(a.x) | ((unsigned)f2bfu(a.y) << 16);
  o.y = (unsigned)f2bfu(a.z) | ((unsigned)f2bfu(a.w) << 16);
  o.z = (unsigned)f2bfu(b.x) | ((unsigned)f2bfu(b.y) << 16);
  o.w = (unsigned)f2bfu(b.z) | ((unsigned)f2bfu(b.w) << 16);
  *reinterpret_cast<uint4*>(cm_bf + i) = o;
}

// ---------------------------------------------------------------------------
// Kernel 3: y[t,c] = sum_{a,p} x[t,a] s[t,p] cm[c,a,p]
// GEMM M=1024, N=256, K=65536 via v_wmma_f32_16x16x32_bf16.
// Workgroup = 8 waves: 16 tokens x 128 channels. Outer loop over a:
//   - 256 threads build As[t,p]=x[t,a]*s[t,p] (16x256 bf16, 8KB) in LDS once
//     via v_pk_mul_bf16 (s held in registers), shared by all 8 waves,
//   - each wave runs 8 unrolled WMMA K-steps fed by ds_load_b128 (A) and
//     global_load_b128 (B, bf16 concept_map, L2-resident).
// ---------------------------------------------------------------------------
template <bool PRECONV>
__global__ void __launch_bounds__(256)
gemm_kernel(const float* __restrict__ cm_f32,
            const unsigned short* __restrict__ cm_bf,
            const unsigned short* __restrict__ x_bf,
            const unsigned short* __restrict__ s_bf,
            float* __restrict__ y) {
  __shared__ uint4 sh_x4[512];   // 16 tokens x 256 bf16 x-values (8KB)
  __shared__ uint4 sh_a4[512];   // 16 tokens x 256 bf16 A-slice   (8KB)

  const int tid   = threadIdx.x;
  const int wave  = tid >> 5;
  const int lane  = tid & 31;
  const int t_blk = blockIdx.x >> 1;                  // 64 token blocks
  const int c_blk = ((blockIdx.x & 1) << 3) + wave;   // 16 channel blocks

  // ---- stage x (bf16) for this token block, and keep each thread's 16
  //      s-values (token tt, p in [pp,pp+16)) resident in registers
  const size_t tok_base = (size_t)t_blk * 16 * E_DIM;
  {
    const uint4* gx = reinterpret_cast<const uint4*>(x_bf + tok_base);
    sh_x4[tid]       = gx[tid];
    sh_x4[tid + 256] = gx[tid + 256];
  }
  const int tt = tid >> 4;            // fill-role token 0..15
  const int pp = (tid & 15) << 4;     // fill-role p base (16 halves)
  uint4 sseg0, sseg1;
  {
    const uint4* gsv = reinterpret_cast<const uint4*>(s_bf + tok_base + tt * E_DIM + pp);
    sseg0 = gsv[0];
    sseg1 = gsv[1];
  }
  __syncthreads();

  const unsigned short* shx = reinterpret_cast<const unsigned short*>(sh_x4);
  const int row = lane & 15;          // A row (token) == B/D column index
  const int hi  = lane >> 4;          // lane-half: K sub-run select (ISA layout)
  const int c   = (c_blk << 4) + row; // output channel

  v8f acc = {};

  for (int a = 0; a < E_DIM; ++a) {
    // ---- cooperative rank-1 fill: As[t,p] = x[t,a] * s[t,p]  (v_pk_mul_bf16)
    {
      unsigned xs = (unsigned)shx[tt * E_DIM + a] * 0x10001u;  // replicate bf16
      uint4 r0, r1;
      r0.x = pk_mul_bf16(sseg0.x, xs); r0.y = pk_mul_bf16(sseg0.y, xs);
      r0.z = pk_mul_bf16(sseg0.z, xs); r0.w = pk_mul_bf16(sseg0.w, xs);
      r1.x = pk_mul_bf16(sseg1.x, xs); r1.y = pk_mul_bf16(sseg1.y, xs);
      r1.z = pk_mul_bf16(sseg1.z, xs); r1.w = pk_mul_bf16(sseg1.w, xs);
      const int ai = tt * 32 + (tid & 15) * 2;   // uint4 index into sh_a4
      sh_a4[ai]     = r0;
      sh_a4[ai + 1] = r1;
    }
    __syncthreads();

    // prefetch next a-slice of concept_map into cache hierarchy
    if (PRECONV && a + 1 < E_DIM)
      __builtin_prefetch(cm_bf + (size_t)c * K_TOT + (size_t)(a + 1) * E_DIM + (lane << 3), 0, 0);

    // ---- 8 WMMA K-steps over p0 = 0,32,...,224
#pragma unroll
    for (int ps = 0; ps < 8; ++ps) {
      // A 16x32 bf16 frag: lane half 'hi' holds K = hi*8+[0..8), 16+hi*8+[0..8)
      const int abase = row * 32 + ps * 4 + hi;   // uint4 units (8 halves each)
      struct U8 { uint4 lo, hiq; } ua{sh_a4[abase], sh_a4[abase + 2]};
      v16us aus = __builtin_bit_cast(v16us, ua);

      v16us bus;
      if constexpr (PRECONV) {
        const unsigned short* bb = cm_bf + (size_t)c * K_TOT + a * E_DIM + ps * 32 + hi * 8;
        U8 ub{*reinterpret_cast<const uint4*>(bb),
              *reinterpret_cast<const uint4*>(bb + 16)};
        bus = __builtin_bit_cast(v16us, ub);
      } else {
        const float* fb = cm_f32 + (size_t)c * K_TOT + a * E_DIM + ps * 32 + hi * 8;
        float4 f0 = *reinterpret_cast<const float4*>(fb);
        float4 f1 = *reinterpret_cast<const float4*>(fb + 4);
        float4 f2 = *reinterpret_cast<const float4*>(fb + 16);
        float4 f3 = *reinterpret_cast<const float4*>(fb + 20);
        bus[0]  = f2bfu(f0.x); bus[1]  = f2bfu(f0.y); bus[2]  = f2bfu(f0.z); bus[3]  = f2bfu(f0.w);
        bus[4]  = f2bfu(f1.x); bus[5]  = f2bfu(f1.y); bus[6]  = f2bfu(f1.z); bus[7]  = f2bfu(f1.w);
        bus[8]  = f2bfu(f2.x); bus[9]  = f2bfu(f2.y); bus[10] = f2bfu(f2.z); bus[11] = f2bfu(f2.w);
        bus[12] = f2bfu(f3.x); bus[13] = f2bfu(f3.y); bus[14] = f2bfu(f3.z); bus[15] = f2bfu(f3.w);
      }

      acc = __builtin_amdgcn_wmma_f32_16x16x32_bf16(
          false, __builtin_bit_cast(v16bf, aus),
          false, __builtin_bit_cast(v16bf, bus),
          (short)0, acc, false, false);
    }
    __syncthreads();   // sh_a consumed before next fill overwrites it
  }

  // D layout: lane n = lane&15 (channel), VGPR r -> row m = r + hi*8
  float* ybase = y + tok_base + c;
#pragma unroll
  for (int r = 0; r < 8; ++r)
    ybase[(size_t)(r + hi * 8) * E_DIM] = acc[r];
}

// ---------------------------------------------------------------------------
// Kernel 4: residual + LayerNorm (eps = 1e-3), one token per block.
// ---------------------------------------------------------------------------
__global__ void ln_kernel(const float* __restrict__ yg,
                          const float* __restrict__ x,
                          const float* __restrict__ gamma,
                          const float* __restrict__ beta,
                          float* __restrict__ out) {
  __shared__ float red[256];
  const int e = threadIdx.x;
  const size_t idx = (size_t)blockIdx.x * E_DIM + e;
  float v = yg[idx] + x[idx];
  red[e] = v;
  __syncthreads();
  for (int off = 128; off > 0; off >>= 1) {
    if (e < off) red[e] += red[e + off];
    __syncthreads();
  }
  float mean = red[0] * (1.0f / E_DIM);
  __syncthreads();
  float d = v - mean;
  red[e] = d * d;
  __syncthreads();
  for (int off = 128; off > 0; off >>= 1) {
    if (e < off) red[e] += red[e + off];
    __syncthreads();
  }
  float var = red[0] * (1.0f / E_DIM);
  out[idx] = d * rsqrtf(var + 1e-3f) * gamma[e] + beta[e];
}

// ---------------------------------------------------------------------------
extern "C" void kernel_launch(void* const* d_in, const int* in_sizes, int n_in,
                              void* d_out, int out_size, void* d_ws, size_t ws_size,
                              hipStream_t stream) {
  (void)in_sizes; (void)n_in; (void)out_size;
  const float* x     = (const float*)d_in[0];
  const float* cm    = (const float*)d_in[1];
  const float* gamma = (const float*)d_in[2];
  const float* beta  = (const float*)d_in[3];
  float* out = (float*)d_out;
  char* ws = (char*)d_ws;

  unsigned short* s_bf  = (unsigned short*)ws;                         // 512 KB
  unsigned short* x_bf  = (unsigned short*)(ws + (512u << 10));        // 512 KB
  float*          y     = (float*)(ws + (1024u << 10));                // 1 MB
  unsigned short* cm_bf = (unsigned short*)(ws + (2048u << 10));       // 32 MB

  prefix_kernel<<<(T_TOK * E_DIM) / 256, 256, 0, stream>>>(x, x_bf, s_bf);

  const size_t need_preconv = ((size_t)2048u << 10) + (size_t)K_TOT * E_DIM * 2;
  if (ws_size >= need_preconv) {
    cvt_cm_kernel<<<(K_TOT / 8) * E_DIM / 256, 256, 0, stream>>>(cm, cm_bf);
    gemm_kernel<true><<<128, 256, 0, stream>>>(cm, cm_bf, x_bf, s_bf, y);
  } else {
    gemm_kernel<false><<<128, 256, 0, stream>>>(cm, cm_bf, x_bf, s_bf, y);
  }

  ln_kernel<<<T_TOK, 256, 0, stream>>>(y, x, gamma, beta, out);
}
